// LookupAlignConvolution2d_55439437856824
// MI455X (gfx1250) — matrix operand order repaired and checked by reading, not compiled, not run on hardware
//
#include <hip/hip_runtime.h>
#include <cstdint>

// ---------------------------------------------------------------------------
// Implicit-GEMM 3x3 VALID conv for gfx1250 (MI455X), f16 WMMA, f32 accumulate,
// B-operand staged by the Tensor Data Mover (tensor_load_to_lds + TENSORcnt).
//   x:      (32, 56, 56, 256) NHWC fp32
//   weight: (256, 256, 3, 3)  OIHW fp32 (|w| < 0.01 -> 0)
//   bias:   (256,) fp32
//   out:    (32, 54, 54, 256) NHWC fp32
// GEMM view: M = 32*54*54 = 93312, N = 256, K = 9*256 = 2304.
// ---------------------------------------------------------------------------

typedef _Float16     v8h  __attribute__((ext_vector_type(8)));
typedef _Float16     v16h __attribute__((ext_vector_type(16)));
typedef float        v8f  __attribute__((ext_vector_type(8)));
typedef unsigned int v4u  __attribute__((ext_vector_type(4)));
typedef int          v4i  __attribute__((ext_vector_type(4)));
typedef int          v8i  __attribute__((ext_vector_type(8)));

#define C_IN   256
#define C_OUT  256
#define H_IN   56
#define W_IN   56
#define OH     54
#define OW     54
#define KTOT   2304     // 9 * 256
#define MTOT   93312    // 32 * 54 * 54
#define BM     128
#define BN     128
#define BK     32
#define PADH   40       // LDS row stride in halves (80B; conflict-free frag loads)
#define KITER  72       // KTOT / BK
#define SPARSE_TH 0.01f

// Use the TDM for B staging when the builtin exists (device pass only).
#if defined(__HIP_DEVICE_COMPILE__) && defined(__has_builtin)
#  if __has_builtin(__builtin_amdgcn_tensor_load_to_lds)
#    define USE_TDM 1
#  endif
#endif
#ifndef USE_TDM
#  define USE_TDM 0
#endif

// k-chunk kt -> fp32-element offset into x row base: (kh*56 + kw)*256 + ci0
__device__ __forceinline__ int tap_off(int kt) {
  int tap = kt >> 3;            // 0..8 (8 ci-chunks of 32 per tap)
  int ci0 = (kt & 7) * 32;
  int kh  = tap / 3;
  int kw  = tap - kh * 3;
  return (kh * W_IN + kw) * C_IN + ci0;
}

__device__ __forceinline__ v8h cvt8(float4 a, float4 b) {
  v8h r;
  r[0] = (_Float16)a.x; r[1] = (_Float16)a.y; r[2] = (_Float16)a.z; r[3] = (_Float16)a.w;
  r[4] = (_Float16)b.x; r[5] = (_Float16)b.y; r[6] = (_Float16)b.z; r[7] = (_Float16)b.w;
  return r;
}

#if USE_TDM
__device__ __forceinline__ void wait_tensor0() {
#if __has_builtin(__builtin_amdgcn_s_wait_tensorcnt)
  __builtin_amdgcn_s_wait_tensorcnt(0);
#else
  asm volatile("s_wait_tensorcnt 0x0" ::: "memory");
#endif
}

// TDM: load one 128(n) x 32(k) f16 tile, row stride 2304 halves, into LDS with
// 16B padding after each 64B row (-> 80B LDS row stride = PADH*2).
// D# per CDNA5 ISA ch.8: group0 = {flags/count, lds_addr, global_addr, type},
// group1 = {mask/size/pad, dims/strides}; groups 2/3 unused (2D tensor).
__device__ __forceinline__ void tdm_load_b_tile(const _Float16* gsrc,
                                                unsigned lds_addr) {
  unsigned long long ga = (unsigned long long)(uintptr_t)gsrc;
  v4u g0 = { 1u,                                   // count=1 (user descriptor)
             lds_addr,                             // lds_addr (bytes)
             (unsigned)ga,                         // global_addr[31:0]
             (unsigned)(ga >> 32) | (2u << 30) };  // global_addr[56:32] | type=2
  v8i g1 = {
      (int)((1u << 16)      // data_size = 1 -> 2 bytes
            | (1u << 20)    // pad_enable
            | (3u << 22)    // pad_interval: 16 DWORDs (64B) between pads
            | (3u << 25)),  // pad_amount:   4 DWORDs (16B)
      (int)(((unsigned)KTOT & 0xFFFFu) << 16),     // tensor_dim0 lo16 (=2304)
      (int)(((unsigned)C_OUT & 0xFFFFu) << 16),    // dim0 hi16=0 | tensor_dim1 lo16 (=256)
      (int)(32u << 16),                            // dim1 hi16=0 | tile_dim0 = 32
      (int)BN,                                     // tile_dim1 = 128, tile_dim2 = 0
      (int)KTOT,                                   // tensor_dim0_stride lo32 (=2304)
      0,                                           // stride hi16 | dim1_stride lo16
      0 };
  v4i gz = { 0, 0, 0, 0 };
#if __clang_major__ >= 23
  v8i gz8 = { 0, 0, 0, 0, 0, 0, 0, 0 };
  __builtin_amdgcn_tensor_load_to_lds(g0, g1, gz, gz, gz8, 0);
#else
  __builtin_amdgcn_tensor_load_to_lds(g0, g1, gz, gz, 0);
#endif
}
#endif  // USE_TDM

// ---------------------------------------------------------------------------
// Kernel 1: mask + convert weights OIHW fp32 -> B^T f16 layout [n][k],
//           k = (kh*3 + kw)*256 + ci  (matches A's k ordering).
// ---------------------------------------------------------------------------
__global__ void prep_weights_kernel(const float* __restrict__ w,
                                    _Float16* __restrict__ wf) {
  int t   = blockIdx.x * 256 + threadIdx.x;  // t = n*2304 + k  (coalesced write)
  int n   = t / KTOT;
  int k   = t - n * KTOT;
  int tap = k >> 8;
  int ci  = k & 255;
  int kh  = tap / 3;
  int kw  = tap - kh * 3;
  float v = w[((n * C_IN + ci) * 3 + kh) * 3 + kw];
  wf[t] = (fabsf(v) < SPARSE_TH) ? (_Float16)0.f : (_Float16)v;
}

// ---------------------------------------------------------------------------
// Kernel 2: implicit-GEMM conv, double-buffered LDS, 8 waves x (32x64) tile.
// ---------------------------------------------------------------------------
__global__ __launch_bounds__(256, 2)
void conv_implicit_gemm_wmma(const float* __restrict__ x,
                             const _Float16* __restrict__ wf,
                             const float* __restrict__ bias,
                             float* __restrict__ out) {
  __shared__ alignas(16) _Float16 Ash[2][BM * PADH];
  __shared__ alignas(16) _Float16 Bsh[2][BN * PADH];

  const int tid  = threadIdx.x;
  const int lane = tid & 31;          // wave32
  const int wv   = tid >> 5;          // 8 waves
  const int wm   = wv & 3;            // wave row (4 x 32 rows)
  const int wn   = wv >> 2;           // wave col (2 x 64 cols)

  const int mBase = (int)(blockIdx.x >> 1) * BM;
  const int nBase = (int)(blockIdx.x & 1)  * BN;

  // --- staging assignment: thread -> (tile row r, 16-element half h) ---
  const int r = tid >> 1;             // 0..127
  const int h = tid & 1;              // 0..1
  int m   = mBase + r;
  int img = m / (OH * OW);
  int rem = m - img * (OH * OW);
  int oh  = rem / OW;
  int ow  = rem - oh * OW;
  const float* xrow = x + ((img * H_IN + oh) * W_IN + ow) * C_IN + h * 16;
  const unsigned sofs = (unsigned)(r * PADH + h * 16);   // halves
#if USE_TDM
  const _Float16* wtile = wf + nBase * KTOT;             // + kt*BK per step
#else
  const _Float16* wrow  = wf + (nBase + r) * KTOT + h * 16;
#endif

  // --- accumulators: 2 (M) x 4 (N) tiles of 16x16 f32 ---
  v8f acc[2][4];
  {
    v8f z = {0.f, 0.f, 0.f, 0.f, 0.f, 0.f, 0.f, 0.f};
#pragma unroll
    for (int i = 0; i < 2; ++i)
#pragma unroll
      for (int j = 0; j < 4; ++j) acc[i][j] = z;
  }

  // fragment-load lane decode (ISA 16-bit A 16x32 / B 32x16 layouts)
  const int ln = lane & 15;
  const int hA = (lane >> 4) * 8;     // A: lane<16 -> halves {0..7,16..23}; else {8..15,24..31}
  const int hB = (lane >> 4) * 8;     // B: lane<16 -> K {0..7,16..23}; else {8..15,24..31}

  float4 af[4];
#if !USE_TDM
  v8h bflo, bfhi;
#endif

  // --- prologue: stage kt = 0 into buffer 0 ---
#if USE_TDM
  if (wv == 0) tdm_load_b_tile(wtile, (unsigned)(uintptr_t)&Bsh[0][0]);
#endif
  {
    const float* p = xrow + tap_off(0);
    af[0] = ((const float4*)p)[0];
    af[1] = ((const float4*)p)[1];
    af[2] = ((const float4*)p)[2];
    af[3] = ((const float4*)p)[3];
    v8h alo = cvt8(af[0], af[1]);
    v8h ahi = cvt8(af[2], af[3]);
    *(v8h*)(&Ash[0][sofs])     = alo;
    *(v8h*)(&Ash[0][sofs + 8]) = ahi;
#if !USE_TDM
    bflo = ((const v8h*)wrow)[0];
    bfhi = ((const v8h*)wrow)[1];
    *(v8h*)(&Bsh[0][sofs])     = bflo;
    *(v8h*)(&Bsh[0][sofs + 8]) = bfhi;
#endif
  }
#if USE_TDM
  if (wv == 0) wait_tensor0();
#endif
  __syncthreads();

  for (int kt = 0; kt < KITER; ++kt) {
    const int buf = kt & 1;

    // issue movement for next K-step (hidden under compute below)
    if (kt + 1 < KITER) {
#if USE_TDM
      if (wv == 0)
        tdm_load_b_tile(wtile + (kt + 1) * BK,
                        (unsigned)(uintptr_t)&Bsh[buf ^ 1][0]);
#endif
      const float* p = xrow + tap_off(kt + 1);
      af[0] = ((const float4*)p)[0];
      af[1] = ((const float4*)p)[1];
      af[2] = ((const float4*)p)[2];
      af[3] = ((const float4*)p)[3];
#if !USE_TDM
      const _Float16* q = wrow + (kt + 1) * BK;
      bflo = ((const v8h*)q)[0];
      bfhi = ((const v8h*)q)[1];
#endif
    }
    if (kt + 2 < KITER) {
      // near-cache (WGP-scope) prefetch: this WGP consumes these lines shortly
      __builtin_prefetch(xrow + tap_off(kt + 2), 0, 3);  // global_prefetch_b8
    }

    // --- compute: 12 ds_load_b128 + 8 v_wmma_f32_16x16x32_f16 per wave ---
    {
      const _Float16* As = Ash[buf];
      const _Float16* Bs = Bsh[buf];
      v16h a[2], b[4];
#pragma unroll
      for (int mt = 0; mt < 2; ++mt) {
        int row = wm * 32 + mt * 16 + ln;
        v8h lo = *(const v8h*)(As + row * PADH + hA);
        v8h hi = *(const v8h*)(As + row * PADH + hA + 16);
        a[mt] = __builtin_shufflevector(lo, hi, 0,1,2,3,4,5,6,7,8,9,10,11,12,13,14,15);
      }
#pragma unroll
      for (int nt = 0; nt < 4; ++nt) {
        int row = wn * 64 + nt * 16 + ln;
        v8h lo = *(const v8h*)(Bs + row * PADH + hB);
        v8h hi = *(const v8h*)(Bs + row * PADH + hB + 16);
        b[nt] = __builtin_shufflevector(lo, hi, 0,1,2,3,4,5,6,7,8,9,10,11,12,13,14,15);
      }
#pragma unroll
      for (int mt = 0; mt < 2; ++mt)
#pragma unroll
        for (int nt = 0; nt < 4; ++nt)
          acc[mt][nt] = __builtin_amdgcn_wmma_f32_16x16x32_f16(
              false, a[mt], false, b[nt], (short)0, acc[mt][nt], false, false);
    }

    // --- publish next K-step ---
    if (kt + 1 < KITER) {
      const int nb = buf ^ 1;
      v8h alo = cvt8(af[0], af[1]);
      v8h ahi = cvt8(af[2], af[3]);
      *(v8h*)(&Ash[nb][sofs])     = alo;
      *(v8h*)(&Ash[nb][sofs + 8]) = ahi;
#if !USE_TDM
      *(v8h*)(&Bsh[nb][sofs])     = bflo;
      *(v8h*)(&Bsh[nb][sofs + 8]) = bfhi;
#else
      if (wv == 0) wait_tensor0();   // B tile DMA complete before publishing
#endif
      __syncthreads();
    }
  }

  // --- epilogue: C/D layout => VGPR rr: lanes0-15 M=rr, lanes16-31 M=rr+8 ---
  // Non-temporal stores: keep the write-once 96MB output stream from evicting
  // the L2-resident x (103MB, re-read across taps/N-blocks) and f16 weights.
#pragma unroll
  for (int nt = 0; nt < 4; ++nt) {
    int n = nBase + wn * 64 + nt * 16 + ln;
    float bv = bias[n];
#pragma unroll
    for (int mt = 0; mt < 2; ++mt) {
      int m0 = mBase + wm * 32 + mt * 16 + ((lane >> 4) << 3);
#pragma unroll
      for (int rr = 0; rr < 8; ++rr) {
        __builtin_nontemporal_store(acc[mt][nt][rr] + bv,
                                    out + (m0 + rr) * C_OUT + n);
      }
    }
  }
}

// ---------------------------------------------------------------------------
// Safety fallback (only if workspace is too small for the f16 weights).
// ---------------------------------------------------------------------------
__global__ void conv_naive(const float* __restrict__ x, const float* __restrict__ w,
                           const float* __restrict__ bias, float* __restrict__ out,
                           int total) {
  int o = blockIdx.x * 256 + threadIdx.x;
  if (o >= total) return;
  int n   = o & 255;
  int m   = o >> 8;
  int img = m / (OH * OW);
  int rem = m - img * (OH * OW);
  int oh  = rem / OW;
  int ow  = rem - oh * OW;
  float s = bias[n];
  for (int kh = 0; kh < 3; ++kh)
    for (int kw = 0; kw < 3; ++kw) {
      const float* xp = x + ((img * H_IN + oh + kh) * W_IN + ow + kw) * C_IN;
      const float* wp = w + (size_t)n * C_IN * 9 + kh * 3 + kw;
      for (int ci = 0; ci < C_IN; ++ci) {
        float wv = wp[ci * 9];
        wv = (fabsf(wv) < SPARSE_TH) ? 0.f : wv;
        s += xp[ci] * wv;
      }
    }
  out[o] = s;
}

// ---------------------------------------------------------------------------
extern "C" void kernel_launch(void* const* d_in, const int* in_sizes, int n_in,
                              void* d_out, int out_size, void* d_ws, size_t ws_size,
                              hipStream_t stream) {
  const float* x    = (const float*)d_in[0];
  const float* w    = (const float*)d_in[1];
  const float* bias = (const float*)d_in[2];
  float* out        = (float*)d_out;

  const size_t wbytes = (size_t)C_OUT * KTOT * sizeof(_Float16);  // 1.13 MiB
  if (ws_size >= wbytes) {
    _Float16* wf = (_Float16*)d_ws;
    prep_weights_kernel<<<(C_OUT * KTOT) / 256, 256, 0, stream>>>(w, wf);
    conv_implicit_gemm_wmma<<<(MTOT / BM) * (C_OUT / BN), 256, 0, stream>>>(
        x, wf, bias, out);
  } else {
    int total = MTOT * C_OUT;
    conv_naive<<<(total + 255) / 256, 256, 0, stream>>>(x, w, bias, out, total);
  }
}
